// ScaledDotProductAttention2_87634512708214
// MI455X (gfx1250) — matrix operand order repaired and checked by reading, not compiled, not run on hardware
//
#include <hip/hip_runtime.h>

// Scaled dot-product attention (mask + softmax + deterministic dropout) for
// MI455X / gfx1250: flash-attention, bf16 WMMA with f32 accumulation.
constexpr int   kB    = 64;
constexpr int   kSq   = 1024;
constexpr int   kSk   = 1024;
constexpr int   kD    = 256;
constexpr int   kDv   = 256;
constexpr float kScale = 0.0625f;      // 1/sqrt(256)
constexpr float kDropP = 0.1f;
constexpr float kKeep  = 1.0f / 0.9f;  // inverted-dropout scale

typedef __attribute__((ext_vector_type(16))) __bf16       bf16x16;
typedef __attribute__((ext_vector_type(8)))  float        f32x8;
typedef __attribute__((ext_vector_type(4)))  float        f32x4;
typedef __attribute__((ext_vector_type(4)))  unsigned int u32x4;

struct Frag2 { u32x4 a, b; };  // 32 bytes == bf16x16

// Pack two f32 -> packed bf16 pair (RNE) ending with a single v_perm_b32:
// result = { hi.bytes[3,2], lo.bytes[3,2] }.
__device__ __forceinline__ unsigned int pk2_bf16(float lo, float hi) {
  unsigned int a = __builtin_bit_cast(unsigned int, lo);
  unsigned int b = __builtin_bit_cast(unsigned int, hi);
  a += 0x7FFFu + ((a >> 16) & 1u);          // round-to-nearest-even
  b += 0x7FFFu + ((b >> 16) & 1u);
  // bytes 0-3 select from src1 (=a), bytes 4-7 from src0 (=b)
  return __builtin_amdgcn_perm(b, a, 0x07060302u);
}
__device__ __forceinline__ unsigned short f2bf(float x) {
  unsigned int a = __builtin_bit_cast(unsigned int, x);
  a += 0x7FFFu + ((a >> 16) & 1u);
  return (unsigned short)(a >> 16);
}
__device__ __forceinline__ u32x4 pack8(f32x4 x, f32x4 y) {
  u32x4 r;
  r.x = pk2_bf16(x.x, x.y); r.y = pk2_bf16(x.z, x.w);
  r.z = pk2_bf16(y.x, y.y); r.w = pk2_bf16(y.z, y.w);
  return r;
}

// A-matrix (16xK) fragment from row-major bf16 LDS: lane holds row = lane&15,
// K runs {0..7,16..23} (lanes 0-15) / {8..15,24..31} (lanes 16-31).
__device__ __forceinline__ bf16x16 ld_afrag(const unsigned short* p, int hi) {
  Frag2 t;
  t.a = *reinterpret_cast<const u32x4*>(p + hi * 8);
  t.b = *reinterpret_cast<const u32x4*>(p + 16 + hi * 8);
  return __builtin_bit_cast(bf16x16, t);
}
// B-matrix (Kx16) fragment from K-contiguous bf16 LDS: lane holds col = lane&15,
// K runs {0..7,16..23} (lanes 0-15) / {8..15,24..31} (lanes 16-31).
__device__ __forceinline__ bf16x16 ld_bfrag(const unsigned short* p, int hi) {
  Frag2 t;
  t.a = *reinterpret_cast<const u32x4*>(p + hi * 8);
  t.b = *reinterpret_cast<const u32x4*>(p + 16 + hi * 8);
  return __builtin_bit_cast(bf16x16, t);
}

__device__ __forceinline__ f32x8 wmma_bf16(bf16x16 a, bf16x16 b, f32x8 c) {
  return __builtin_amdgcn_wmma_f32_16x16x32_bf16(false, a, false, b,
                                                 (short)0, c, false, false);
}

__global__ __launch_bounds__(256, 1)
void sdpa_fa_kernel(const float* __restrict__ Q, const float* __restrict__ K,
                    const float* __restrict__ V, const int* __restrict__ M,
                    const float* __restrict__ U, float* __restrict__ O) {
  extern __shared__ __align__(16) unsigned short smem[];
  unsigned short* const ldsQ = smem;                 // [8][16][256] bf16 (64 KB)
  unsigned short* const ldsK = ldsQ + 8 * 16 * 256;  // [32][256]   row-major (16 KB)
  unsigned short* const ldsV = ldsK + 32 * 256;      // [256][32]   transposed (16 KB)
  unsigned short* const ldsP = ldsV + 256 * 32;      // [8][16][32] per-wave P (8 KB)

  const int tid  = threadIdx.x;
  const int wave = tid >> 5;
  const int lane = tid & 31;
  const int hi16 = lane >> 4;   // which half of the wave
  const int l16  = lane & 15;

  const int b  = blockIdx.x >> 3;
  const int qw = (blockIdx.x & 7) * 128 + wave * 16;  // this wave's 16 q rows

  // ---- stage this wave's Q tile (16x256) into LDS as bf16 (no barrier needed:
  //      only this wave touches this region, DS ops are in-order per wave) ----
  {
    const float* src = Q + ((size_t)b * kSq + qw + l16) * kD + hi16 * 128;
    unsigned short* dst = ldsQ + (wave * 16 + l16) * 256 + hi16 * 128;
#pragma unroll
    for (int i = 0; i < 16; ++i) {
      f32x4 x = reinterpret_cast<const f32x4*>(src)[2 * i];
      f32x4 y = reinterpret_cast<const f32x4*>(src)[2 * i + 1];
      reinterpret_cast<u32x4*>(dst)[i] = pack8(x, y);
    }
  }

  const f32x8 vzero = {0.f, 0.f, 0.f, 0.f, 0.f, 0.f, 0.f, 0.f};
  f32x8 acc[16];                          // 16x256 f32 output accumulator
#pragma unroll
  for (int i = 0; i < 16; ++i) acc[i] = vzero;
  float mrow[8], lrow[8];                 // running max / sum per row
#pragma unroll
  for (int r = 0; r < 8; ++r) { mrow[r] = -__builtin_inff(); lrow[r] = 0.f; }

  unsigned short* const ldsPw = ldsP + wave * (16 * 32);

  for (int kt = 0; kt < kSk / 32; ++kt) {
    const int kbase = kt * 32;
    __syncthreads();  // previous tile's LDS reads done before re-staging

    // ---- cooperative stage: K tile 32x256 -> bf16 row-major ----
    {
      const int key = tid >> 3, cg = tid & 7;
      const float* src = K + ((size_t)b * kSk + kbase + key) * kD + cg * 32;
      unsigned short* dst = ldsK + key * 256 + cg * 32;
#pragma unroll
      for (int i = 0; i < 4; ++i) {
        f32x4 x = reinterpret_cast<const f32x4*>(src)[2 * i];
        f32x4 y = reinterpret_cast<const f32x4*>(src)[2 * i + 1];
        reinterpret_cast<u32x4*>(dst)[i] = pack8(x, y);
      }
    }
    // ---- cooperative stage: V tile 32x256 -> bf16 transposed [col][key] ----
    {
      const int kp = tid >> 4, cg = tid & 15;  // key-pair, 16-col group
      const float* s0 = V + ((size_t)b * kSk + kbase + 2 * kp) * kDv + cg * 16;
      const float* s1 = s0 + kDv;
      float a0[16], a1[16];
#pragma unroll
      for (int i = 0; i < 4; ++i) {
        f32x4 x = reinterpret_cast<const f32x4*>(s0)[i];
        f32x4 y = reinterpret_cast<const f32x4*>(s1)[i];
        a0[4*i+0]=x.x; a0[4*i+1]=x.y; a0[4*i+2]=x.z; a0[4*i+3]=x.w;
        a1[4*i+0]=y.x; a1[4*i+1]=y.y; a1[4*i+2]=y.z; a1[4*i+3]=y.w;
      }
      unsigned int* vdst = reinterpret_cast<unsigned int*>(ldsV);
#pragma unroll
      for (int j = 0; j < 16; ++j)
        vdst[(cg * 16 + j) * 16 + kp] = pk2_bf16(a0[j], a1[j]);
    }
    if (kt + 1 < kSk / 32) {  // hint next tile into cache
      __builtin_prefetch(K + ((size_t)b * kSk + kbase + 32 + (tid >> 3)) * kD + (tid & 7) * 32, 0, 1);
      __builtin_prefetch(V + ((size_t)b * kSk + kbase + 32 + (tid >> 4) * 2) * kDv + (tid & 15) * 16, 0, 1);
    }
    __syncthreads();

    // ---- S(16x32) = Q(16x256) @ K^T : 16 bf16 WMMAs, f32 accum ----
    f32x8 s0 = vzero, s1 = vzero;
#pragma unroll
    for (int dc = 0; dc < 8; ++dc) {
      bf16x16 aQ = ld_afrag(ldsQ + (wave * 16 + l16) * 256 + dc * 32, hi16);
      bf16x16 b0 = ld_bfrag(ldsK + l16 * 256 + dc * 32, hi16);        // keys 0..15
      bf16x16 b1 = ld_bfrag(ldsK + (16 + l16) * 256 + dc * 32, hi16); // keys 16..31
      s0 = wmma_bf16(aQ, b0, s0);
      s1 = wmma_bf16(aQ, b1, s1);
    }

    // additive padding mask for this lane's two key columns
    const float mb0 = M[(size_t)b * kSk + kbase + l16]      ? 0.f : -1e9f;
    const float mb1 = M[(size_t)b * kSk + kbase + 16 + l16] ? 0.f : -1e9f;

    // ---- online softmax + dropout; pack P -> LDS (bf16) ----
#pragma unroll
    for (int r = 0; r < 8; ++r) {
      float x0 = s0[r] * kScale + mb0;
      float x1 = s1[r] * kScale + mb1;
      float mx = fmaxf(x0, x1);                  // row max across 16 lanes of half
      mx = fmaxf(mx, __shfl_xor(mx, 1));
      mx = fmaxf(mx, __shfl_xor(mx, 2));
      mx = fmaxf(mx, __shfl_xor(mx, 4));
      mx = fmaxf(mx, __shfl_xor(mx, 8));
      const float mn   = fmaxf(mrow[r], mx);
      const float corr = __expf(mrow[r] - mn);
      mrow[r] = mn;
      float p0 = __expf(x0 - mn);
      float p1 = __expf(x1 - mn);
      float rs = p0 + p1;                        // row sum BEFORE dropout
      rs += __shfl_xor(rs, 1);
      rs += __shfl_xor(rs, 2);
      rs += __shfl_xor(rs, 4);
      rs += __shfl_xor(rs, 8);
      lrow[r] = lrow[r] * corr + rs;
#pragma unroll
      for (int dg = 0; dg < 16; ++dg) acc[dg][r] *= corr;  // rescale prior accum
      // deterministic inverted dropout on the numerator
      const size_t drow = ((size_t)b * kSq + qw + r + 8 * hi16) * kSk + kbase;
      p0 *= (U[drow + l16]      >= kDropP) ? kKeep : 0.f;
      p1 *= (U[drow + 16 + l16] >= kDropP) ? kKeep : 0.f;
      ldsPw[(r + 8 * hi16) * 32 + l16]      = f2bf(p0);
      ldsPw[(r + 8 * hi16) * 32 + 16 + l16] = f2bf(p1);
    }

    // ---- acc(16x256) += P(16x32) @ V(32x256) : 16 bf16 WMMAs ----
    bf16x16 aP = ld_afrag(ldsPw + l16 * 32, hi16);
#pragma unroll
    for (int dg = 0; dg < 16; ++dg) {
      bf16x16 bV = ld_bfrag(ldsV + (dg * 16 + l16) * 32, hi16);
      acc[dg] = wmma_bf16(aP, bV, acc[dg]);
    }
  }

  // ---- epilogue: softmax normalization via v_rcp_f32 + mul (avoids 128 IEEE
  //      division expansions), store f32 coalesced ----
  float rinv[8];
#pragma unroll
  for (int r = 0; r < 8; ++r) rinv[r] = __builtin_amdgcn_rcpf(lrow[r]);
#pragma unroll
  for (int dg = 0; dg < 16; ++dg) {
#pragma unroll
    for (int r = 0; r < 8; ++r) {
      O[((size_t)b * kSq + qw + r + 8 * hi16) * kDv + dg * 16 + l16] =
          acc[dg][r] * rinv[r];
    }
  }
}

extern "C" void kernel_launch(void* const* d_in, const int* in_sizes, int n_in,
                              void* d_out, int out_size, void* d_ws, size_t ws_size,
                              hipStream_t stream) {
  (void)in_sizes; (void)n_in; (void)out_size; (void)d_ws; (void)ws_size;
  const float* q    = (const float*)d_in[0];
  const float* k    = (const float*)d_in[1];
  const float* v    = (const float*)d_in[2];
  const int*   mask = (const int*)d_in[3];
  const float* du   = (const float*)d_in[4];
  float*       out  = (float*)d_out;

  const size_t smemBytes =
      (size_t)(8 * 16 * 256 + 32 * 256 + 256 * 32 + 8 * 16 * 32) *
      sizeof(unsigned short);  // 104 KB dynamic LDS (WGP supports 320 KB)
  dim3 grid(kB * (kSq / 128));  // 512 blocks
  dim3 block(256);              // 8 wave32 waves
  hipLaunchKernelGGL(sdpa_fa_kernel, grid, block, smemBytes, stream,
                     q, k, v, mask, du, out);
}
